// CQAttentionLayer_22067541966858
// MI455X (gfx1250) — matrix operand order repaired and checked by reading, not compiled, not run on hardware
//
#include <hip/hip_runtime.h>

// CQAttention for MI455X (gfx1250, wave32, WMMA bf16 split-precision path).
// - b2 = (s1 s1^T) ctx reassociated as s1 (s1^T ctx): the (B,2048,2048) "l"
//   tensor is never materialized (86 GFLOP -> 8.6 GFLOP).
// - All GEMMs: v_wmma_f32_16x16x32_bf16 with hi/lo bf16 splitting
//   (3 WMMAs per product ~ f32 accuracy), f32 accumulate.
// - All WMMA fragments load per-lane contiguous (global_load_b128 clauses):
//   operands are pre-transposed in the prep pass; k_out stages the s1 strip
//   through LDS (transposed, hi/lo packed u32) shared by all 8 waves.
// - hi/lo unpack uses v_perm_b32 (2 words -> 1 packed bf16 pair).

typedef __attribute__((ext_vector_type(16))) __bf16 v16bf;
typedef __attribute__((ext_vector_type(8)))  __bf16 v8bf;
typedef __attribute__((ext_vector_type(8)))  float  v8f;
typedef __attribute__((ext_vector_type(4)))  float  v4f;
typedef __attribute__((ext_vector_type(4)))  unsigned int v4u;
typedef __attribute__((ext_vector_type(8)))  unsigned int v8u;

constexpr int B = 16, D = 128, C = 2048, Q = 512;
constexpr int DOUT = 4 * D;

#define DEV static __device__ __forceinline__

DEV v8f wmma_bf16(v16bf a, v16bf b, v8f c) {
  return __builtin_amdgcn_wmma_f32_16x16x32_bf16(false, a, false, b, (short)0, c,
                                                 false, false);
}

DEV void bsplit(float x, __bf16* h, __bf16* l) {
  __bf16 hh = (__bf16)x;
  *h = hh;
  *l = (__bf16)(x - (float)hh);
}

DEV unsigned int pack_bf2(__bf16 h, __bf16 l) {
  return (unsigned int)__builtin_bit_cast(unsigned short, h) |
         ((unsigned int)__builtin_bit_cast(unsigned short, l) << 16);
}

// Word w = {hi in bytes 0-1, lo in bytes 2-3}. From word pair (a=even, b=odd):
// perm(b, a, 0x05040100) = packed {hi_a, hi_b}; 0x07060302 = {lo_a, lo_b}.
DEV void perm_pair(unsigned int a, unsigned int b, unsigned int* hp,
                   unsigned int* lp) {
  *hp = __builtin_amdgcn_perm(b, a, 0x05040100u);
  *lp = __builtin_amdgcn_perm(b, a, 0x07060302u);
}

// Unpack 16 contiguous packed words into hi/lo v16bf fragments (k_out LDS row).
DEV void unpack_row16(const unsigned int* __restrict__ row, v16bf* h, v16bf* l) {
  v4u u[4];
#pragma unroll
  for (int i = 0; i < 4; ++i) u[i] = *(const v4u*)(row + 4 * i);
  v8u H, L;
#pragma unroll
  for (int p = 0; p < 8; ++p) {
    unsigned int a = u[(2 * p) / 4][(2 * p) % 4];
    unsigned int b = u[(2 * p + 1) / 4][(2 * p + 1) % 4];
    unsigned int hp, lp;
    perm_pair(a, b, &hp, &lp);
    H[p] = hp; L[p] = lp;
  }
  *h = __builtin_bit_cast(v16bf, H);
  *l = __builtin_bit_cast(v16bf, L);
}

// 16-wide bf16 A-fragment slice for one lane from a row-major K-row:
// elements 0..15 -> K = k0+16*half+e (contiguous).
DEV v16bf load_arow(const __bf16* __restrict__ p, int k0, int half) {
  v8bf x0 = *(const v8bf*)(p + k0 + 16 * half);
  v8bf x1 = *(const v8bf*)(p + k0 + 16 * half + 8);
  v16bf r;
#pragma unroll
  for (int e = 0; e < 8; ++e) { r[e] = x0[e]; r[8 + e] = x1[e]; }
  return r;
}

// ---- prep: split context, and transpose (ctx*wcq) to [B][C][D] -----------
__global__ void __launch_bounds__(256) k_prep_ctx(
    const float* __restrict__ context, const float* __restrict__ w,
    __bf16* __restrict__ xh, __bf16* __restrict__ xl,      // ctx     [B][D][C]
    __bf16* __restrict__ yTh, __bf16* __restrict__ yTl) {  // ctx*wcq [B][C][D]
  constexpr int DT = D / 64, CT = C / 64;   // 2, 32
  const int b  = blockIdx.x / (DT * CT);
  const int r  = blockIdx.x % (DT * CT);
  const int d0 = (r / CT) * 64;
  const int c0 = (r % CT) * 64;
  const int t = threadIdx.x;
  __shared__ __align__(16) unsigned int lds[64 * 65];
  {
    const int dl = t >> 2, cs = (t & 3) * 16;
    const int d = d0 + dl;
    const float wv = w[2 * D + d];
    const float* src = context + ((size_t)b * D + d) * C + c0 + cs;
    v4f f[4];
#pragma unroll
    for (int i = 0; i < 4; ++i) f[i] = *(const v4f*)(src + 4 * i);
    v16bf hv, lv;
#pragma unroll
    for (int j = 0; j < 16; ++j) {
      float cv = f[j / 4][j % 4];
      __bf16 h, l; bsplit(cv, &h, &l);
      hv[j] = h; lv[j] = l;
      __bf16 th_, tl_; bsplit(cv * wv, &th_, &tl_);
      lds[dl * 65 + cs + j] = pack_bf2(th_, tl_);
    }
    size_t o = ((size_t)b * D + d) * C + c0 + cs;
    *(v16bf*)(xh + o) = hv;
    *(v16bf*)(xl + o) = lv;
  }
  __syncthreads();
  {
    const int cl = t >> 2, dsg = (t & 3) * 16;
    v8u H, L;
#pragma unroll
    for (int p = 0; p < 8; ++p) {
      unsigned int a = lds[(dsg + 2 * p) * 65 + cl];
      unsigned int b_ = lds[(dsg + 2 * p + 1) * 65 + cl];
      unsigned int hp, lp;
      perm_pair(a, b_, &hp, &lp);
      H[p] = hp; L[p] = lp;
    }
    size_t o = ((size_t)b * C + c0 + cl) * D + d0 + dsg;
    *(v16bf*)(yTh + o) = __builtin_bit_cast(v16bf, H);
    *(v16bf*)(yTl + o) = __builtin_bit_cast(v16bf, L);
  }
}

// ---- prep: split query (straight) and transposed [B][Q][D] ---------------
__global__ void __launch_bounds__(256) k_prep_qry(
    const float* __restrict__ query,
    __bf16* __restrict__ xh, __bf16* __restrict__ xl,      // qry  [B][D][Q]
    __bf16* __restrict__ yTh, __bf16* __restrict__ yTl) {  // qryT [B][Q][D]
  constexpr int DT = D / 64, QT = Q / 64;   // 2, 8
  const int b  = blockIdx.x / (DT * QT);
  const int r  = blockIdx.x % (DT * QT);
  const int d0 = (r / QT) * 64;
  const int q0 = (r % QT) * 64;
  const int t = threadIdx.x;
  __shared__ __align__(16) unsigned int lds[64 * 65];
  {
    const int dl = t >> 2, qs = (t & 3) * 16;
    const float* src = query + ((size_t)b * D + d0 + dl) * Q + q0 + qs;
    v4f f[4];
#pragma unroll
    for (int i = 0; i < 4; ++i) f[i] = *(const v4f*)(src + 4 * i);
    v16bf hv, lv;
#pragma unroll
    for (int j = 0; j < 16; ++j) {
      __bf16 h, l; bsplit(f[j / 4][j % 4], &h, &l);
      hv[j] = h; lv[j] = l;
      lds[dl * 65 + qs + j] = pack_bf2(h, l);
    }
    size_t o = ((size_t)b * D + d0 + dl) * Q + q0 + qs;
    *(v16bf*)(xh + o) = hv;
    *(v16bf*)(xl + o) = lv;
  }
  __syncthreads();
  {
    const int ql_ = t >> 2, dsg = (t & 3) * 16;
    v8u H, L;
#pragma unroll
    for (int p = 0; p < 8; ++p) {
      unsigned int a = lds[(dsg + 2 * p) * 65 + ql_];
      unsigned int b_ = lds[(dsg + 2 * p + 1) * 65 + ql_];
      unsigned int hp, lp;
      perm_pair(a, b_, &hp, &lp);
      H[p] = hp; L[p] = lp;
    }
    size_t o = ((size_t)b * Q + q0 + ql_) * D + d0 + dsg;
    *(v16bf*)(yTh + o) = __builtin_bit_cast(v16bf, H);
    *(v16bf*)(yTl + o) = __builtin_bit_cast(v16bf, L);
  }
}

// ---- bias vectors: cw[b,c] = ctx@wc, qw[b,q] = qry@wq --------------------
__global__ void __launch_bounds__(256) k_vec(
    const float* __restrict__ context, const float* __restrict__ query,
    const float* __restrict__ w, float* __restrict__ cw, float* __restrict__ qw) {
  int idx = blockIdx.x * 256 + threadIdx.x;
  if (idx < B * C) {
    int b = idx / C, c = idx % C;
    const float* p = context + (size_t)b * D * C + c;
    float s = 0.f;
    for (int d = 0; d < D; ++d) s += p[(size_t)d * C] * w[D + d];
    cw[idx] = s;
  } else {
    int j = idx - B * C;
    int b = j / Q, q = j % Q;
    const float* p = query + (size_t)b * D * Q + q;
    float s = 0.f;
    for (int d = 0; d < D; ++d) s += p[(size_t)d * Q] * w[d];
    qw[j] = s;
  }
}

// ---- score GEMM: s[b,q,c] = qw + cw + sum_d qryT*(ctx*wcq)^T -------------
__global__ void __launch_bounds__(256) k_score(
    const __bf16* __restrict__ qTh, const __bf16* __restrict__ qTl,  // [B][Q][D]
    const __bf16* __restrict__ aTh, const __bf16* __restrict__ aTl,  // [B][C][D]
    const float* __restrict__ cw, const float* __restrict__ qw,
    __bf16* __restrict__ sh, __bf16* __restrict__ sl) {              // [B][Q][C]
  const int lane = threadIdx.x & 31;
  const int tile = blockIdx.x * 8 + (threadIdx.x >> 5);
  constexpr int QT = Q / 16, CT = C / 64;   // 32, 32
  const int b  = tile / (QT * CT);
  const int r  = tile % (QT * CT);
  const int q0 = (r / CT) * 16;
  const int c0 = (r % CT) * 64;
  const int M = lane & 15, half = lane >> 4;
  const __bf16* arh = qTh + ((size_t)b * Q + q0 + M) * D;
  const __bf16* arl = qTl + ((size_t)b * Q + q0 + M) * D;
  size_t brow[4];
#pragma unroll
  for (int n = 0; n < 4; ++n) brow[n] = ((size_t)b * C + c0 + n * 16 + M) * D;

  v8f acc[4] = {};
  for (int k0 = 0; k0 < D; k0 += 32) {
    v16bf Ah = load_arow(arh, k0, half);
    v16bf Al = load_arow(arl, k0, half);
#pragma unroll
    for (int n = 0; n < 4; ++n) {
      v16bf Bh = *(const v16bf*)(aTh + brow[n] + k0 + 16 * half);
      v16bf Bl = *(const v16bf*)(aTl + brow[n] + k0 + 16 * half);
      acc[n] = wmma_bf16(Ah, Bh, acc[n]);
      acc[n] = wmma_bf16(Ah, Bl, acc[n]);
      acc[n] = wmma_bf16(Al, Bh, acc[n]);
    }
  }
#pragma unroll
  for (int n = 0; n < 4; ++n) {
    const int c = c0 + n * 16 + M;
    const float cwv = cw[b * C + c];
#pragma unroll
    for (int rr = 0; rr < 8; ++rr) {
      const int q = q0 + rr + 8 * half;
      float v = acc[n][rr] + cwv + qw[b * Q + q];
      size_t o = ((size_t)b * Q + q) * C + c;
      bsplit(v, &sh[o], &sl[o]);
    }
  }
}

// ---- softmax over C, in place on the bf16 hi/lo pair ---------------------
__global__ void __launch_bounds__(256) k_softmax(__bf16* __restrict__ sh,
                                                 __bf16* __restrict__ sl) {
  const int row = blockIdx.x;   // b*Q + q
  const int tid = threadIdx.x;
  __bf16* ph = sh + (size_t)row * C;
  __bf16* pl = sl + (size_t)row * C;
  __shared__ float red[256];
  float x[8];
  float m = -3.402823466e38f;
#pragma unroll
  for (int i = 0; i < 8; ++i) {
    int idx = tid + i * 256;
    x[i] = (float)ph[idx] + (float)pl[idx];
    m = fmaxf(m, x[i]);
  }
  red[tid] = m; __syncthreads();
  for (int s = 128; s > 0; s >>= 1) {
    if (tid < s) red[tid] = fmaxf(red[tid], red[tid + s]);
    __syncthreads();
  }
  m = red[0]; __syncthreads();
  float sum = 0.f;
#pragma unroll
  for (int i = 0; i < 8; ++i) { x[i] = __expf(x[i] - m); sum += x[i]; }
  red[tid] = sum; __syncthreads();
  for (int s = 128; s > 0; s >>= 1) {
    if (tid < s) red[tid] += red[tid + s];
    __syncthreads();
  }
  const float inv = 1.f / red[0];
#pragma unroll
  for (int i = 0; i < 8; ++i) {
    int idx = tid + i * 256;
    bsplit(x[i] * inv, &ph[idx], &pl[idx]);
  }
}

// ---- t GEMM: t[b,d,q] = sum_c ctx[b,d,c]*s1[b,q,c]  (M=d,N=q,K=c) --------
__global__ void __launch_bounds__(256) k_tgemm(
    const __bf16* __restrict__ ch, const __bf16* __restrict__ cl,   // ctx [B][D][C]
    const __bf16* __restrict__ sh, const __bf16* __restrict__ sl,   // s1 [B][Q][C]
    __bf16* __restrict__ th, __bf16* __restrict__ tl) {             // t  [B][D][Q]
  const int lane = threadIdx.x & 31;
  const int tile = blockIdx.x * 8 + (threadIdx.x >> 5);
  constexpr int DT = D / 16, QT = Q / 16;   // 8, 32
  const int b  = tile / (DT * QT);
  const int r  = tile % (DT * QT);
  const int d0 = (r / QT) * 16;
  const int q0 = (r % QT) * 16;
  const int M = lane & 15, half = lane >> 4;
  const __bf16* arh = ch + (size_t)b * D * C + (size_t)(d0 + M) * C;
  const __bf16* arl = cl + (size_t)b * D * C + (size_t)(d0 + M) * C;
  const size_t srow = ((size_t)b * Q + (q0 + M)) * C;

  v8f acc = {};
  for (int k0 = 0; k0 < C; k0 += 32) {
    v16bf Ah = load_arow(arh, k0, half);
    v16bf Al = load_arow(arl, k0, half);
    v16bf Bh = *(const v16bf*)(sh + srow + k0 + 16 * half);
    v16bf Bl = *(const v16bf*)(sl + srow + k0 + 16 * half);
    acc = wmma_bf16(Ah, Bh, acc);
    acc = wmma_bf16(Ah, Bl, acc);
    acc = wmma_bf16(Al, Bh, acc);
  }
#pragma unroll
  for (int rr = 0; rr < 8; ++rr) {
    const int d = d0 + rr + 8 * half;
    size_t o = ((size_t)b * D + d) * Q + (q0 + M);
    bsplit(acc[rr], &th[o], &tl[o]);
  }
}

// ---- fused final GEMMs + epilogue ---------------------------------------
// Block = 8 waves = all of D for one (b, 64-wide c strip). Per K-step the
// 32x64 s1 strip is staged once into LDS (transposed, hi/lo packed u32) and
// shared by all waves; fragments unpack via v_perm_b32 pairs.
__global__ void __launch_bounds__(256) k_out(
    const __bf16* __restrict__ qh, const __bf16* __restrict__ ql,   // qry [B][D][Q]
    const __bf16* __restrict__ th, const __bf16* __restrict__ tl,   // t   [B][D][Q]
    const __bf16* __restrict__ sh, const __bf16* __restrict__ sl,   // s1  [B][Q][C]
    const float* __restrict__ context, float* __restrict__ out) {
  constexpr int CT = C / 64;   // 32
  const int b  = blockIdx.x / CT;
  const int c0 = (blockIdx.x % CT) * 64;
  const int lane = threadIdx.x & 31;
  const int d0 = (threadIdx.x >> 5) * 16;     // 8 waves cover D=128
  const int M = lane & 15, half = lane >> 4;

  __shared__ __align__(16) unsigned int lds[64 * 36];   // [c_loc][k] {hi,lo}

  const size_t arow = ((size_t)b * D + d0 + M) * Q;
  const int kl  = threadIdx.x >> 3;           // staging k row 0..31
  const int csg = (threadIdx.x & 7) * 8;      // staging c segment
  const size_t srow_base = (size_t)b * Q * C + c0 + csg;

  v8f accA[4] = {};
  v8f accB[4] = {};
  for (int k0 = 0; k0 < Q; k0 += 32) {
    {  // cooperative transpose-stage of s1 strip
      size_t o = srow_base + (size_t)(k0 + kl) * C;
      v8bf hv = *(const v8bf*)(sh + o);
      v8bf lv = *(const v8bf*)(sl + o);
      int kn = (k0 + 32 < Q) ? 32 : 0;        // prefetch next strip
      __builtin_prefetch((const void*)(sh + o + (size_t)kn * C), 0, 0);
#pragma unroll
      for (int j = 0; j < 8; ++j)
        lds[(csg + j) * 36 + kl] = pack_bf2(hv[j], lv[j]);
    }
    __syncthreads();

    v16bf Aqh = load_arow(qh + arow, k0, half);
    v16bf Aql = load_arow(ql + arow, k0, half);
    v16bf Ath = load_arow(th + arow, k0, half);
    v16bf Atl = load_arow(tl + arow, k0, half);
#pragma unroll
    for (int n = 0; n < 4; ++n) {
      v16bf Bh, Bl;
      unpack_row16(lds + (n * 16 + M) * 36 + 16 * half, &Bh, &Bl);
      accA[n] = wmma_bf16(Aqh, Bh, accA[n]);
      accA[n] = wmma_bf16(Aqh, Bl, accA[n]);
      accA[n] = wmma_bf16(Aql, Bh, accA[n]);
      accB[n] = wmma_bf16(Ath, Bh, accB[n]);
      accB[n] = wmma_bf16(Ath, Bl, accB[n]);
      accB[n] = wmma_bf16(Atl, Bh, accB[n]);
    }
    __syncthreads();
  }
  const size_t obase = (size_t)b * DOUT * C;
#pragma unroll
  for (int n = 0; n < 4; ++n) {
    const int c = c0 + n * 16 + M;
#pragma unroll
    for (int rr = 0; rr < 8; ++rr) {
      const int d = d0 + rr + 8 * half;
      const float ctxv = context[((size_t)b * D + d) * C + c];
      const float av = accA[n][rr];
      const float b2v = accB[n][rr];
      out[obase + (size_t)d * C + c] = ctxv;
      out[obase + (size_t)(D + d) * C + c] = av;
      out[obase + (size_t)(2 * D + d) * C + c] = ctxv * av;
      out[obase + (size_t)(3 * D + d) * C + c] = ctxv * b2v;
    }
  }
}

extern "C" void kernel_launch(void* const* d_in, const int* in_sizes, int n_in,
                              void* d_out, int out_size, void* d_ws, size_t ws_size,
                              hipStream_t stream) {
  (void)in_sizes; (void)n_in; (void)out_size; (void)ws_size;
  const float* context = (const float*)d_in[0];
  const float* query   = (const float*)d_in[1];
  const float* w       = (const float*)d_in[2];
  float* out = (float*)d_out;

  char* ws = (char*)d_ws;
  size_t off = 0;
  auto take = [&](size_t bytes) -> void* {
    void* p = ws + off;
    off += (bytes + 255) & ~(size_t)255;
    return p;
  };
  __bf16* ctx_h  = (__bf16*)take((size_t)B * D * C * 2);
  __bf16* ctx_l  = (__bf16*)take((size_t)B * D * C * 2);
  __bf16* cwqT_h = (__bf16*)take((size_t)B * C * D * 2);
  __bf16* cwqT_l = (__bf16*)take((size_t)B * C * D * 2);
  __bf16* qry_h  = (__bf16*)take((size_t)B * D * Q * 2);
  __bf16* qry_l  = (__bf16*)take((size_t)B * D * Q * 2);
  __bf16* qryT_h = (__bf16*)take((size_t)B * Q * D * 2);
  __bf16* qryT_l = (__bf16*)take((size_t)B * Q * D * 2);
  __bf16* s_h    = (__bf16*)take((size_t)B * Q * C * 2);
  __bf16* s_l    = (__bf16*)take((size_t)B * Q * C * 2);
  __bf16* t_h    = (__bf16*)take((size_t)B * D * Q * 2);
  __bf16* t_l    = (__bf16*)take((size_t)B * D * Q * 2);
  float*  cw     = (float*)take((size_t)B * C * 4);
  float*  qw     = (float*)take((size_t)B * Q * 4);

  k_prep_ctx<<<B * (D / 64) * (C / 64), 256, 0, stream>>>(
      context, w, ctx_h, ctx_l, cwqT_h, cwqT_l);
  k_prep_qry<<<B * (D / 64) * (Q / 64), 256, 0, stream>>>(
      query, qry_h, qry_l, qryT_h, qryT_l);
  k_vec<<<(B * C + B * Q) / 256, 256, 0, stream>>>(context, query, w, cw, qw);
  k_score<<<(B * (Q / 16) * (C / 64)) / 8, 256, 0, stream>>>(
      qryT_h, qryT_l, cwqT_h, cwqT_l, cw, qw, s_h, s_l);
  k_softmax<<<B * Q, 256, 0, stream>>>(s_h, s_l);
  k_tgemm<<<(B * (D / 16) * (Q / 16)) / 8, 256, 0, stream>>>(
      ctx_h, ctx_l, s_h, s_l, t_h, t_l);
  k_out<<<B * (C / 64), 256, 0, stream>>>(
      qry_h, qry_l, t_h, t_l, s_h, s_l, context, out);
}
